// HGAT_50998441672758
// MI455X (gfx1250) — matrix operand order repaired and checked by reading, not compiled, not run on hardware
//
#include <hip/hip_runtime.h>
#include <hip/hip_bf16.h>

#define N_NODES   50000
#define T_STEPS   128
#define IN_F      6
#define HID       32
#define G3        96
#define C_OUT     32
#define R_OUT     16
#define NUM_EDGES 2000
#define N_INC     150000

typedef __attribute__((ext_vector_type(16))) __bf16 v16bf;
typedef __attribute__((ext_vector_type(8)))  float  v8f;

static __device__ __forceinline__ __bf16 tobf(float x) { return (__bf16)x; }

static __device__ __forceinline__ v8f vsplat8(float x) {
    v8f r;
#pragma unroll
    for (int i = 0; i < 8; ++i) r[i] = x;
    return r;
}

static __device__ __forceinline__ v16bf bzero16() {
    v16bf r;
#pragma unroll
    for (int i = 0; i < 16; ++i) r[i] = (__bf16)0.0f;
    return r;
}

// Branchless transcendentals: CDNA5 native v_tanh_f32 (TRANS pipe,
// co-executes with XDL WMMA). Fallback: v_exp + v_rcp, still branchless.
#if __has_builtin(__builtin_amdgcn_tanhf)
static __device__ __forceinline__ float fast_tanh(float x) {
    return __builtin_amdgcn_tanhf(x);
}
#else
static __device__ __forceinline__ float fast_tanh(float x) {
    float e = __expf(2.0f * x);
    return 1.0f - 2.0f * __builtin_amdgcn_rcpf(e + 1.0f);
}
#endif

static __device__ __forceinline__ v8f wmma_bf16(v16bf a, v16bf b, v8f c) {
    // D = A(16x32) * B(32x16) + C, f32 accumulate
    return __builtin_amdgcn_wmma_f32_16x16x32_bf16(false, a, false, b, (short)0, c, false, false);
}

#define LDSH 34  // padded row stride (even -> 8B aligned b64 LDS loads)

// ---------------------------------------------------------------------------
// GRU: one wave per 16-node tile, 8 waves / block (128 nodes / block).
// h recurrence on v_wmma_f32_16x16x32_bf16, per-step LDS transpose D->A layout.
// r/z-gate weight fragments are pre-scaled by 0.5 so sigmoid(x) is exactly
// fma(v_tanh(wmma_out), 0.5, 0.5) -- no pre-scale muls in the serial loop.
// ---------------------------------------------------------------------------
__global__ __launch_bounds__(256) void gru_kernel(
    const float* __restrict__ X,      // [N, T, IN_F]
    const float* __restrict__ Wih,    // [96, 6]
    const float* __restrict__ Whh,    // [96, 32]
    const float* __restrict__ bih,    // [96]
    const float* __restrict__ bhh,    // [96]
    float* __restrict__ out,          // [N, 32], leaky_relu(0.01) applied
    int n)
{
    __shared__ float s_Whh[G3 * HID];     // 3072
    __shared__ float s_Wih[G3 * IN_F];    // 576
    __shared__ float s_bih[G3];
    __shared__ float s_bhh[G3];
    __shared__ float s_h[8][16 * LDSH];   // per-wave transpose tile

    const int tid  = threadIdx.x;
    const int lane = tid & 31;
    const int wv   = tid >> 5;
    const int col  = lane & 15;   // N-index / M-row within 16
    const int grp  = lane >> 4;   // lane half

    for (int i = tid; i < G3 * HID; i += 256) s_Whh[i] = Whh[i];
    for (int i = tid; i < G3 * IN_F; i += 256) s_Wih[i] = Wih[i];
    for (int i = tid; i < G3; i += 256) { s_bih[i] = bih[i]; s_bhh[i] = bhh[i]; }
    __syncthreads();

    // --- Build B fragments (contiguous K order: lane%16 = N col, (lane/16)*16+2v+h = K)
    // Chunks 0..3 (r,z gates) pre-scaled by 0.5 for the tanh-sigmoid identity.
    v16bf fWhh[6], fWih[6];
#pragma unroll
    for (int c = 0; c < 6; ++c) {
        const float sc = (c < 4) ? 0.5f : 1.0f;
        const int g = c * 16 + col;
#pragma unroll
        for (int v = 0; v < 8; ++v) {
            const int K = grp * 16 + 2 * v;
            fWhh[c][2 * v]     = tobf(sc * s_Whh[g * HID + K]);
            fWhh[c][2 * v + 1] = tobf(sc * s_Whh[g * HID + K + 1]);
            fWih[c][2 * v]     = (K     < IN_F) ? tobf(sc * s_Wih[g * IN_F + K])     : (__bf16)0.0f;
            fWih[c][2 * v + 1] = (K + 1 < IN_F) ? tobf(sc * s_Wih[g * IN_F + K + 1]) : (__bf16)0.0f;
        }
    }
    // --- Hoisted C-operand bias vectors (WMMA does not destroy C)
    const v8f cb0 = vsplat8(0.5f * (s_bih[0 * 16 + col] + s_bhh[0 * 16 + col]));
    const v8f cb1 = vsplat8(0.5f * (s_bih[1 * 16 + col] + s_bhh[1 * 16 + col]));
    const v8f cb2 = vsplat8(0.5f * (s_bih[2 * 16 + col] + s_bhh[2 * 16 + col]));
    const v8f cb3 = vsplat8(0.5f * (s_bih[3 * 16 + col] + s_bhh[3 * 16 + col]));
    const v8f cbh0 = vsplat8(s_bhh[4 * 16 + col]);
    const v8f cbh1 = vsplat8(s_bhh[5 * 16 + col]);
    const v8f cbi0 = vsplat8(s_bih[4 * 16 + col]);
    const v8f cbi1 = vsplat8(s_bih[5 * 16 + col]);

    const int tile = blockIdx.x * 128 + wv * 16;
    int nd = tile + col; if (nd > n - 1) nd = n - 1;
    const float* xp = X + (long)nd * T_STEPS * IN_F;

    // h state: D-layout (lane = col, reg = row) + A-layout bf16 fragment
    v8f hD0 = vsplat8(0.0f), hD1 = vsplat8(0.0f);
    v16bf hA = bzero16();
    float* hrow = &s_h[wv][0];
    const float xmask = (grp == 0) ? 1.0f : 0.0f;

    for (int t = 0; t < T_STEPS; ++t) {
        // ---- input fragment: x_t padded K=6 -> 32 (upper lane-half all zero)
        const float2* px = (const float2*)(xp + t * IN_F);
        float2 a0 = px[0], a1 = px[1], a2 = px[2];
        __builtin_prefetch(xp + (t + 8) * IN_F, 0, 0);  // speculative, row is 3KB contiguous
        v16bf xf = bzero16();
        xf[0] = tobf(a0.x * xmask); xf[1] = tobf(a0.y * xmask);
        xf[2] = tobf(a1.x * xmask); xf[3] = tobf(a1.y * xmask);
        xf[4] = tobf(a2.x * xmask); xf[5] = tobf(a2.y * xmask);

        // ---- gates: r (chunks 0,1), z (2,3) fused (pre-scaled); n (4,5) split
        v8f gr0 = wmma_bf16(xf, fWih[0], wmma_bf16(hA, fWhh[0], cb0));
        v8f gr1 = wmma_bf16(xf, fWih[1], wmma_bf16(hA, fWhh[1], cb1));
        v8f gz0 = wmma_bf16(xf, fWih[2], wmma_bf16(hA, fWhh[2], cb2));
        v8f gz1 = wmma_bf16(xf, fWih[3], wmma_bf16(hA, fWhh[3], cb3));
        v8f gh0 = wmma_bf16(hA, fWhh[4], cbh0);
        v8f gh1 = wmma_bf16(hA, fWhh[5], cbh1);
        v8f gi0 = wmma_bf16(xf, fWih[4], cbi0);
        v8f gi1 = wmma_bf16(xf, fWih[5], cbi1);

#pragma unroll
        for (int i = 0; i < 8; ++i) {
            float r0 = fmaf(fast_tanh(gr0[i]), 0.5f, 0.5f);
            float z0 = fmaf(fast_tanh(gz0[i]), 0.5f, 0.5f);
            float n0 = fast_tanh(fmaf(r0, gh0[i], gi0[i]));
            hD0[i] = fmaf(z0, hD0[i] - n0, n0);
            float r1 = fmaf(fast_tanh(gr1[i]), 0.5f, 0.5f);
            float z1 = fmaf(fast_tanh(gz1[i]), 0.5f, 0.5f);
            float n1 = fast_tanh(fmaf(r1, gh1[i], gi1[i]));
            hD1[i] = fmaf(z1, hD1[i] - n1, n1);
        }

        // ---- transpose D-layout -> A-layout through per-wave LDS tile
#pragma unroll
        for (int i = 0; i < 8; ++i) {
            const int mrow = i + 8 * grp;
            hrow[mrow * LDSH + col]      = hD0[i];
            hrow[mrow * LDSH + 16 + col] = hD1[i];
        }
#pragma unroll
        for (int v = 0; v < 8; ++v) {
            const int K0 = grp * 16 + 2 * v;
            float2 hv = *(const float2*)&hrow[col * LDSH + K0];
            hA[2 * v]     = tobf(hv.x);
            hA[2 * v + 1] = tobf(hv.y);
        }
    }

    // ---- store leaky_relu(h, 0.01)
#pragma unroll
    for (int i = 0; i < 8; ++i) {
        const int row = tile + i + 8 * grp;
        if (row < n) {
            float v0 = hD0[i]; v0 = v0 > 0.0f ? v0 : 0.01f * v0;
            float v1 = hD1[i]; v1 = v1 > 0.0f ? v1 : 0.01f * v1;
            out[(long)row * HID + col]      = v0;
            out[(long)row * HID + 16 + col] = v1;
        }
    }
}

// ---------------------------------------------------------------------------
// Dense y = x @ W^T (+bias, leaky) via WMMA. x: [n,32], W: [outF,32].
// ---------------------------------------------------------------------------
__global__ __launch_bounds__(256) void dense_wmma(
    const float* __restrict__ x, const float* __restrict__ W,
    const float* __restrict__ bias, float* __restrict__ out,
    int n, int outF, float slope, int doact)
{
    const int lane = threadIdx.x & 31;
    const int wv   = threadIdx.x >> 5;
    const int col  = lane & 15;
    const int grp  = lane >> 4;
    const int tile = blockIdx.x * 128 + wv * 16;

    int r = tile + col; if (r > n - 1) r = n - 1;
    v16bf af;
#pragma unroll
    for (int v = 0; v < 8; ++v) {
        const int K0 = grp * 16 + 2 * v;
        float2 a = *(const float2*)&x[(long)r * HID + K0];
        af[2 * v] = tobf(a.x); af[2 * v + 1] = tobf(a.y);
    }

    const int nch = outF >> 4;
    for (int c = 0; c < nch; ++c) {
        const int g = c * 16 + col;
        v16bf bf_;
#pragma unroll
        for (int v = 0; v < 8; ++v) {
            const int K0 = grp * 16 + 2 * v;
            float2 w = *(const float2*)&W[(long)g * HID + K0];
            bf_[2 * v] = tobf(w.x); bf_[2 * v + 1] = tobf(w.y);
        }
        v8f acc = wmma_bf16(af, bf_, vsplat8(0.0f));
#pragma unroll
        for (int i = 0; i < 8; ++i) {
            const int row = tile + i + 8 * grp;
            if (row < n) {
                float v = acc[i];
                if (doact) {
                    v += bias[c * 16 + col];
                    v = v > 0.0f ? v : slope * v;
                }
                out[(long)row * outF + c * 16 + col] = v;
            }
        }
    }
}

// ---------------------------------------------------------------------------
// Hypergraph helpers
// ---------------------------------------------------------------------------
__global__ void zero_kernel(float* p, int n) {
    int i = blockIdx.x * blockDim.x + threadIdx.x;
    if (i < n) p[i] = 0.0f;
}

__global__ void deg_kernel(const int* __restrict__ node_idx, const int* __restrict__ edge_idx,
                           float* Dcnt, float* Bcnt, int n) {
    int i = blockIdx.x * blockDim.x + threadIdx.x;
    if (i < n) {
        atomicAdd(&Dcnt[node_idx[i]], 1.0f);
        atomicAdd(&Bcnt[edge_idx[i]], 1.0f);
    }
}

__global__ void scatter_edge(const float* __restrict__ xw, const int* __restrict__ node_idx,
                             const int* __restrict__ edge_idx, float* ef, int ninc) {
    int k = blockIdx.x * blockDim.x + threadIdx.x;
    if (k < ninc * 32) {
        int i = k >> 5, f = k & 31;
        atomicAdd(&ef[edge_idx[i] * 32 + f], xw[(long)node_idx[i] * 32 + f]);
    }
}

__global__ void scale_edges(float* ef, const float* __restrict__ Bcnt, int ne) {
    int k = blockIdx.x * blockDim.x + threadIdx.x;
    if (k < ne * 32) {
        float b = Bcnt[k >> 5];
        ef[k] *= (b > 0.0f ? __builtin_amdgcn_rcpf(b) : 0.0f);
    }
}

__global__ void gather_node(const float* __restrict__ ef, const int* __restrict__ node_idx,
                            const int* __restrict__ edge_idx, float* acc, int ninc) {
    int k = blockIdx.x * blockDim.x + threadIdx.x;
    if (k < ninc * 32) {
        int i = k >> 5, f = k & 31;
        atomicAdd(&acc[(long)node_idx[i] * 32 + f], ef[edge_idx[i] * 32 + f]);
    }
}

__global__ void finalize_conv(float* acc, const float* __restrict__ Dcnt,
                              const float* __restrict__ b, int n, float slope) {
    int k = blockIdx.x * blockDim.x + threadIdx.x;
    if (k < n * 32) {
        int nd = k >> 5, f = k & 31;
        float d = Dcnt[nd];
        float v = acc[k] * (d > 0.0f ? __builtin_amdgcn_rcpf(d) : 0.0f) + b[f];
        acc[k] = v > 0.0f ? v : slope * v;
    }
}

// ---------------------------------------------------------------------------
extern "C" void kernel_launch(void* const* d_in, const int* in_sizes, int n_in,
                              void* d_out, int out_size, void* d_ws, size_t ws_size,
                              hipStream_t stream) {
    const float* X    = (const float*)d_in[0];
    const float* Wih  = (const float*)d_in[2];
    const float* Whh  = (const float*)d_in[3];
    const float* bih  = (const float*)d_in[4];
    const float* bhh  = (const float*)d_in[5];
    const float* W1   = (const float*)d_in[6];
    const float* b1   = (const float*)d_in[7];
    const float* W2   = (const float*)d_in[8];
    const float* b2   = (const float*)d_in[9];
    const float* Wl   = (const float*)d_in[10];
    const float* bl   = (const float*)d_in[11];
    const int* node_idx = (const int*)d_in[12];
    const int* edge_idx = (const int*)d_in[13];
    float* outp = (float*)d_out;

    float* buf0 = (float*)d_ws;                 // N*32
    float* buf1 = buf0 + (long)N_NODES * 32;    // N*32
    float* buf2 = buf1 + (long)N_NODES * 32;    // N*32
    float* ef   = buf2 + (long)N_NODES * 32;    // 2000*32
    float* Bcnt = ef + NUM_EDGES * 32;          // 2000
    float* Dcnt = Bcnt + NUM_EDGES;             // 50000

    const int TB = 256;
    const int gb_nodes = (N_NODES + 127) / 128;           // wmma tiles
    const int gb_inc32 = (N_INC * 32 + TB - 1) / TB;
    const int gb_n32   = (N_NODES * 32 + TB - 1) / TB;

    // degrees (B and D), shared by both conv layers
    zero_kernel<<<(NUM_EDGES + N_NODES + TB - 1) / TB, TB, 0, stream>>>(Bcnt, NUM_EDGES + N_NODES);
    deg_kernel<<<(N_INC + TB - 1) / TB, TB, 0, stream>>>(node_idx, edge_idx, Dcnt, Bcnt, N_INC);

    // GRU -> buf0 (leaky 0.01)
    gru_kernel<<<gb_nodes, TB, 0, stream>>>(X, Wih, Whh, bih, bhh, buf0, N_NODES);

    // ---- conv layer 1 ----
    dense_wmma<<<gb_nodes, TB, 0, stream>>>(buf0, W1, nullptr, buf1, N_NODES, 32, 0.0f, 0);
    zero_kernel<<<(NUM_EDGES * 32 + TB - 1) / TB, TB, 0, stream>>>(ef, NUM_EDGES * 32);
    zero_kernel<<<gb_n32, TB, 0, stream>>>(buf2, N_NODES * 32);
    scatter_edge<<<gb_inc32, TB, 0, stream>>>(buf1, node_idx, edge_idx, ef, N_INC);
    scale_edges<<<(NUM_EDGES * 32 + TB - 1) / TB, TB, 0, stream>>>(ef, Bcnt, NUM_EDGES);
    gather_node<<<gb_inc32, TB, 0, stream>>>(ef, node_idx, edge_idx, buf2, N_INC);
    finalize_conv<<<gb_n32, TB, 0, stream>>>(buf2, Dcnt, b1, N_NODES, 0.2f);

    // ---- conv layer 2 ----
    dense_wmma<<<gb_nodes, TB, 0, stream>>>(buf2, W2, nullptr, buf1, N_NODES, 32, 0.0f, 0);
    zero_kernel<<<(NUM_EDGES * 32 + TB - 1) / TB, TB, 0, stream>>>(ef, NUM_EDGES * 32);
    zero_kernel<<<gb_n32, TB, 0, stream>>>(buf0, N_NODES * 32);
    scatter_edge<<<gb_inc32, TB, 0, stream>>>(buf1, node_idx, edge_idx, ef, N_INC);
    scale_edges<<<(NUM_EDGES * 32 + TB - 1) / TB, TB, 0, stream>>>(ef, Bcnt, NUM_EDGES);
    gather_node<<<gb_inc32, TB, 0, stream>>>(ef, node_idx, edge_idx, buf0, N_INC);
    finalize_conv<<<gb_n32, TB, 0, stream>>>(buf0, Dcnt, b2, N_NODES, 0.2f);

    // ---- final linear: leaky(x @ Wl^T + bl, 0.01) -> d_out [N,16]
    dense_wmma<<<gb_nodes, TB, 0, stream>>>(buf0, Wl, bl, outp, N_NODES, R_OUT, 0.01f, 1);
    (void)in_sizes; (void)n_in; (void)out_size; (void)ws_size;
}
